// StructuralContrastiveLoss_90091234001463
// MI455X (gfx1250) — matrix operand order repaired and checked by reading, not compiled, not run on hardware
//
#include <hip/hip_runtime.h>
#include <hip/hip_bf16.h>

typedef __attribute__((ext_vector_type(16))) _Float16 v16h;
typedef __attribute__((ext_vector_type(8)))  _Float16 v8h;
typedef __attribute__((ext_vector_type(8)))  float    v8f;

#define B_SIZE 4096
#define S_SLOTS 8
#define D_DIM 256
#define INV_TEMP 14.285714285714286f   /* 1 / 0.07 */
#define LAMBDA_W 0.3f
#define COL_CHUNKS 4
#define COLS_PER_CHUNK (B_SIZE / COL_CHUNKS)   /* 1024 */

// ---------------------------------------------------------------------------
// Kernel 1: mean-pool over S slots + L2 normalize, emit f16 rows.
// One 256-thread block per row; thread d owns dim d.
// ---------------------------------------------------------------------------
__device__ __forceinline__ void pool_norm_row(const float* __restrict__ src,
                                              _Float16* __restrict__ dst,
                                              int row, int d, int lane, int wave,
                                              float* red)
{
    const float* p = src + (size_t)row * S_SLOTS * D_DIM + d;
    float v = 0.f;
    #pragma unroll
    for (int s = 0; s < S_SLOTS; ++s) v += p[s * D_DIM];
    v *= (1.0f / S_SLOTS);

    float ss = v * v;
    #pragma unroll
    for (int m = 16; m >= 1; m >>= 1) ss += __shfl_xor(ss, m, 32);
    if (lane == 0) red[wave] = ss;
    __syncthreads();
    float tot = 0.f;
    #pragma unroll
    for (int w = 0; w < 8; ++w) tot += red[w];
    const float scale = 1.0f / fmaxf(sqrtf(tot), 1e-12f);
    dst[(size_t)row * D_DIM + d] = (_Float16)(v * scale);
    __syncthreads();
}

__global__ void pool_norm_kernel(const float* __restrict__ xi,
                                 const float* __restrict__ xj,
                                 _Float16* __restrict__ riH,
                                 _Float16* __restrict__ rjH)
{
    __shared__ float red[8];
    const int row  = blockIdx.x;
    const int d    = threadIdx.x;
    const int lane = d & 31;
    const int wave = d >> 5;
    pool_norm_row(xi, riH, row, d, lane, wave, red);
    pool_norm_row(xj, rjH, row, d, lane, wave, red);
}

// ---------------------------------------------------------------------------
// Kernel 2 helpers
// ---------------------------------------------------------------------------
__device__ __forceinline__ void load_btile(v16h b[8], const _Float16* brow)
{
    // B in this code's contiguous k order (matches A): lane n = lane%16 holds 16 contiguous K halves at kb+g*16.
    #pragma unroll
    for (int k = 0; k < 8; ++k)
        b[k] = *(const v16h*)(brow + k * 32);
}

__device__ __forceinline__ void compute_tile(const v16h a[8], const v16h b[8],
                                             int col0, int n16, int g, int row0,
                                             const int* __restrict__ labels,
                                             float expAcc[8], float cntAcc[8],
                                             float simAcc[8])
{
    v8f c = {};
    #pragma unroll
    for (int k = 0; k < 8; ++k)
        c = __builtin_amdgcn_wmma_f32_16x16x32_f16(
                /*neg_a=*/false, a[k], /*neg_b=*/false, b[k],
                /*c_mod=*/(short)0, c, /*reuse_a=*/false, /*reuse_b=*/false);

    // C layout: lane holds N = lane%16; VGPR r holds M = r + 8*g.
    const int gcol = col0 + n16;
    const int* lab = labels + gcol;
    #pragma unroll
    for (int r = 0; r < 8; ++r) {
        const int grow  = row0 + r + g * 8;
        const float s   = c[r] * INV_TEMP;
        const bool diag = (grow == gcol);
        // labels are a 67MB single-pass stream: non-temporal to preserve L2
        // residency of the hot rjH panels.
        const int lv    = __builtin_nontemporal_load(lab + (size_t)grow * B_SIZE);
        const bool pos  = (lv != 0) && !diag;
        expAcc[r] += diag ? 0.0f : __expf(s);
        cntAcc[r] += pos ? 1.0f : 0.0f;
        simAcc[r] += pos ? s : 0.0f;
    }
}

// ---------------------------------------------------------------------------
// Kernel 2: fused sim GEMM (WMMA f16 -> f32) + masked exp/row reductions.
// Block = 256 threads (8 waves). Wave w owns rows [blockIdx.x*128 + w*16, +16)
// and columns [blockIdx.y*1024, +1024). A fragments stay in VGPRs; B tiles
// are double-buffered in VGPRs so loads for tile t+1 overlap WMMAs of tile t.
// ---------------------------------------------------------------------------
__global__ void __launch_bounds__(256, 1)
sim_tile_kernel(const _Float16* __restrict__ riH,
                const _Float16* __restrict__ rjH,
                const int* __restrict__ labels,
                float* __restrict__ expPart,
                float* __restrict__ cntPart,
                float* __restrict__ simPart)
{
    const int tid  = threadIdx.x;
    const int lane = tid & 31;
    const int wave = tid >> 5;
    const int g    = lane >> 4;    // half-wave id (ISA A/B/C layout split)
    const int n16  = lane & 15;

    const int row0    = blockIdx.x * 128 + wave * 16;
    const int chunk   = blockIdx.y;
    const int colBase = chunk * COLS_PER_CHUNK;

    // ---- Preload A fragments: rows row0..row0+15, full K=256 (8 x 16x32) ----
    // A in this code's contiguous k order (matches B): lane m = lane%16 holds 16 contiguous K halves at kb+g*16.
    v16h a[8];
    {
        const _Float16* arow = riH + (size_t)(row0 + n16) * D_DIM;
        #pragma unroll
        for (int k = 0; k < 8; ++k) {
            const int kb = k * 32;
            a[k] = *(const v16h*)(arow + kb + g * 16);
        }
    }

    float expAcc[8], cntAcc[8], simAcc[8];
    #pragma unroll
    for (int r = 0; r < 8; ++r) { expAcc[r] = 0.f; cntAcc[r] = 0.f; simAcc[r] = 0.f; }

    const _Float16* bbase = rjH + (size_t)(colBase + n16) * D_DIM + (size_t)g * 16;
    const size_t tileStride = (size_t)16 * D_DIM;   // halves per 16-col tile
    constexpr int T = COLS_PER_CHUNK / 16;          // 64 tiles, even

    v16h b0[8], b1[8];
    load_btile(b0, bbase);

    for (int t = 0; t < T; t += 2) {
        // prefetch tile t+1 while tile t computes
        load_btile(b1, bbase + (size_t)(t + 1) * tileStride);
        compute_tile(a, b0, colBase + t * 16, n16, g, row0,
                     labels, expAcc, cntAcc, simAcc);
        // prefetch tile t+2 while tile t+1 computes
        if (t + 2 < T)
            load_btile(b0, bbase + (size_t)(t + 2) * tileStride);
        compute_tile(a, b1, colBase + (t + 1) * 16, n16, g, row0,
                     labels, expAcc, cntAcc, simAcc);
    }

    // Reduce across the 16 lanes of each half (xor masks stay within a half).
    #pragma unroll
    for (int r = 0; r < 8; ++r) {
        #pragma unroll
        for (int m = 1; m <= 8; m <<= 1) {
            expAcc[r] += __shfl_xor(expAcc[r], m, 32);
            cntAcc[r] += __shfl_xor(cntAcc[r], m, 32);
            simAcc[r] += __shfl_xor(simAcc[r], m, 32);
        }
    }
    if (n16 == 0) {  // lane 0 -> rows row0+0..7, lane 16 -> rows row0+8..15
        #pragma unroll
        for (int r = 0; r < 8; ++r) {
            const int grow   = row0 + r + g * 8;
            const size_t idx = (size_t)chunk * B_SIZE + grow;
            expPart[idx] = expAcc[r];
            cntPart[idx] = cntAcc[r];
            simPart[idx] = simAcc[r];
        }
    }
}

// ---------------------------------------------------------------------------
// Kernel 3: single-block deterministic finalize -> scalar loss.
// ---------------------------------------------------------------------------
__global__ void __launch_bounds__(1024)
finalize_kernel(const float* __restrict__ expPart,
                const float* __restrict__ cntPart,
                const float* __restrict__ simPart,
                float* __restrict__ out)
{
    __shared__ float sTot[32], sCnt[32];
    const int tid = threadIdx.x;
    float tot = 0.f, cnt = 0.f;
    for (int row = tid; row < B_SIZE; row += 1024) {
        float e = 0.f, c = 0.f, sm = 0.f;
        #pragma unroll
        for (int p = 0; p < COL_CHUNKS; ++p) {
            e  += expPart[p * B_SIZE + row];
            c  += cntPart[p * B_SIZE + row];
            sm += simPart[p * B_SIZE + row];
        }
        const float logd = __logf(e + 1e-8f);
        tot += c * logd - sm;
        cnt += c;
    }
    #pragma unroll
    for (int m = 16; m >= 1; m >>= 1) {
        tot += __shfl_xor(tot, m, 32);
        cnt += __shfl_xor(cnt, m, 32);
    }
    if ((tid & 31) == 0) { sTot[tid >> 5] = tot; sCnt[tid >> 5] = cnt; }
    __syncthreads();
    if (tid == 0) {
        float T = 0.f, C = 0.f;
        for (int w = 0; w < 32; ++w) { T += sTot[w]; C += sCnt[w]; }
        const float loss = (C > 0.f) ? (T / fmaxf(C, 1.f)) : 0.f;
        out[0] = LAMBDA_W * loss;
    }
}

// ---------------------------------------------------------------------------
extern "C" void kernel_launch(void* const* d_in, const int* in_sizes, int n_in,
                              void* d_out, int out_size, void* d_ws, size_t ws_size,
                              hipStream_t stream) {
    const float* xi     = (const float*)d_in[0];
    const float* xj     = (const float*)d_in[1];
    const int*   labels = (const int*)d_in[2];

    char* ws = (char*)d_ws;
    _Float16* riH    = (_Float16*)(ws);                               // 2 MB
    _Float16* rjH    = (_Float16*)(ws + (size_t)2 * 1024 * 1024);     // 2 MB
    float* expPart   = (float*)(ws + (size_t)4 * 1024 * 1024);        // 64 KB
    float* cntPart   = (float*)(ws + (size_t)4 * 1024 * 1024 + 64 * 1024);
    float* simPart   = (float*)(ws + (size_t)4 * 1024 * 1024 + 128 * 1024);

    pool_norm_kernel<<<B_SIZE, 256, 0, stream>>>(xi, xj, riH, rjH);

    dim3 grid2(B_SIZE / 128, COL_CHUNKS);
    sim_tile_kernel<<<grid2, 256, 0, stream>>>(riH, rjH, labels,
                                               expPart, cntPart, simPart);

    finalize_kernel<<<1, 1024, 0, stream>>>(expPart, cntPart, simPart,
                                            (float*)d_out);
}
